// QuantLlamaAttention_67783173865538
// MI455X (gfx1250) — matrix operand order repaired and checked by reading, not compiled, not run on hardware
//
#include <hip/hip_runtime.h>
#include <hip/hip_bf16.h>

typedef __attribute__((ext_vector_type(16))) _Float16 v16h;
typedef __attribute__((ext_vector_type(8)))  _Float16 v8h;
typedef __attribute__((ext_vector_type(8)))  float    v8f;

#define S_LEN 2048
#define E_DIM 2048
#define N_HEAD 32
#define HEAD_D 64

// A-operand fragment (16x32 f16): lane holds row m=lane%16,
// contiguous k = 16*(lane/16) + s (matches B operand order).
__device__ __forceinline__ v16h load_a_frag(const _Float16* __restrict__ arow, int k0, int half) {
    v8h lo = *(const v8h*)(arow + k0 + half * 16);
    v8h hi = *(const v8h*)(arow + k0 + half * 16 + 8);
    return __builtin_shufflevector(lo, hi, 0,1,2,3,4,5,6,7,8,9,10,11,12,13,14,15);
}

__device__ __forceinline__ v8f wmma_f16(v16h a, v16h b, v8f c) {
    return __builtin_amdgcn_wmma_f32_16x16x32_f16(false, a, false, b, (short)0, c, false, false);
}

// ---------------------------------------------------------------------------
// 1) fp32 -> f16 elementwise convert (hidden states)
// ---------------------------------------------------------------------------
__global__ void cvt_f16_kernel(const float* __restrict__ x, _Float16* __restrict__ y, int n) {
    int i = blockIdx.x * blockDim.x + threadIdx.x;
    if (i < n) y[i] = (_Float16)x[i];
}

// ---------------------------------------------------------------------------
// 2) per-output-row int8 fake-quant of a [2048,2048] weight, store dequantized f16
// ---------------------------------------------------------------------------
__global__ void wquant_kernel(const float* __restrict__ W, _Float16* __restrict__ Wq) {
    __shared__ float red[256];
    int row = blockIdx.x;
    const float* wr = W + (size_t)row * E_DIM;
    float m = 0.f;
    for (int c = threadIdx.x; c < E_DIM; c += 256) m = fmaxf(m, fabsf(wr[c]));
    red[threadIdx.x] = m;
    __syncthreads();
    for (int s = 128; s > 0; s >>= 1) {
        if (threadIdx.x < s) red[threadIdx.x] = fmaxf(red[threadIdx.x], red[threadIdx.x + s]);
        __syncthreads();
    }
    float scale = fmaxf(red[0] / 127.0f, 1e-8f);
    _Float16* dq = Wq + (size_t)row * E_DIM;
    for (int c = threadIdx.x; c < E_DIM; c += 256) {
        float q = rintf(wr[c] / scale);
        q = fminf(fmaxf(q, -128.f), 127.f);
        dq[c] = (_Float16)(q * scale);
    }
}

// ---------------------------------------------------------------------------
// 3) WMMA GEMM: C[m,n] = alpha * sum_k A[m,k] * Bt[n,k]   (M=N=K=2048)
//    32(M) x 64(N) per wave, 8 accumulators. Ping-pong fragment sets with a
//    2x-unrolled K loop: no register-rotation moves, loads of one set always
//    in flight while the other set feeds 8 WMMAs.
// ---------------------------------------------------------------------------
struct GFrag { v16h a0, a1, b0, b1, b2, b3; };

__device__ __forceinline__ void gemm_load(GFrag& f,
        const _Float16* __restrict__ arow0, const _Float16* __restrict__ arow1,
        const _Float16* __restrict__ b0p, const _Float16* __restrict__ b1p,
        const _Float16* __restrict__ b2p, const _Float16* __restrict__ b3p,
        int k0, int half) {
    f.a0 = load_a_frag(arow0, k0, half);
    f.a1 = load_a_frag(arow1, k0, half);
    f.b0 = *(const v16h*)(b0p + k0);
    f.b1 = *(const v16h*)(b1p + k0);
    f.b2 = *(const v16h*)(b2p + k0);
    f.b3 = *(const v16h*)(b3p + k0);
}

__device__ __forceinline__ void gemm_compute(const GFrag& f, v8f acc[8]) {
    acc[0] = wmma_f16(f.a0, f.b0, acc[0]);
    acc[4] = wmma_f16(f.a1, f.b0, acc[4]);
    acc[1] = wmma_f16(f.a0, f.b1, acc[1]);
    acc[5] = wmma_f16(f.a1, f.b1, acc[5]);
    acc[2] = wmma_f16(f.a0, f.b2, acc[2]);
    acc[6] = wmma_f16(f.a1, f.b2, acc[6]);
    acc[3] = wmma_f16(f.a0, f.b3, acc[3]);
    acc[7] = wmma_f16(f.a1, f.b3, acc[7]);
}

__global__ void __launch_bounds__(128) gemm_wmma_kernel(
        const _Float16* __restrict__ A, const _Float16* __restrict__ Bt,
        float* __restrict__ C, float alpha) {
    const int K = E_DIM, N = E_DIM;
    int wave = (blockIdx.x * blockDim.x + threadIdx.x) >> 5;
    int lane = threadIdx.x & 31;
    int half = lane >> 4;
    int ln   = lane & 15;
    int nWave = wave & 31;    // 32 waves tile N=2048 (64 each)
    int mTile = wave >> 5;    // 64 M tiles of 32 rows
    int n0 = nWave * 64;
    const _Float16* arow0 = A + (size_t)(mTile * 32 + ln) * K;
    const _Float16* arow1 = arow0 + (size_t)16 * K;
    const _Float16* b0p = Bt + (size_t)(n0 + 0 * 16 + ln) * K + half * 16;
    const _Float16* b1p = Bt + (size_t)(n0 + 1 * 16 + ln) * K + half * 16;
    const _Float16* b2p = Bt + (size_t)(n0 + 2 * 16 + ln) * K + half * 16;
    const _Float16* b3p = Bt + (size_t)(n0 + 3 * 16 + ln) * K + half * 16;

    v8f acc[8] = {};
    GFrag f0, f1;
    gemm_load(f0, arow0, arow1, b0p, b1p, b2p, b3p, 0, half);
    for (int k0 = 0; k0 < K; k0 += 64) {
        gemm_load(f1, arow0, arow1, b0p, b1p, b2p, b3p, k0 + 32, half);
        gemm_compute(f0, acc);
        int kn = (k0 + 64 < K) ? k0 + 64 : K - 32;   // clamped; last-iter load unused
        gemm_load(f0, arow0, arow1, b0p, b1p, b2p, b3p, kn, half);
        gemm_compute(f1, acc);
    }

#pragma unroll
    for (int mt = 0; mt < 2; ++mt)
#pragma unroll
        for (int t = 0; t < 4; ++t)
#pragma unroll
            for (int r = 0; r < 8; ++r) {
                int mm = mTile * 32 + mt * 16 + r + half * 8;  // C/D layout
                int nn = n0 + t * 16 + ln;
                C[(size_t)mm * N + nn] = acc[mt * 4 + t][r] * alpha;
            }
}

// ---------------------------------------------------------------------------
// 4) per-tensor absmax (for activation fake-quant scales)
// ---------------------------------------------------------------------------
__global__ void absmax_kernel(const float* __restrict__ x, int n, unsigned* __restrict__ amax) {
    __shared__ float red[256];
    int stride = gridDim.x * blockDim.x;
    float m = 0.f;
    for (int i = blockIdx.x * blockDim.x + threadIdx.x; i < n; i += stride) {
        if (i + stride < n) __builtin_prefetch(x + i + stride, 0, 1);  // global_prefetch_b8
        m = fmaxf(m, fabsf(x[i]));
    }
    red[threadIdx.x] = m;
    __syncthreads();
    for (int s = 128; s > 0; s >>= 1) {
        if (threadIdx.x < s) red[threadIdx.x] = fmaxf(red[threadIdx.x], red[threadIdx.x + s]);
        __syncthreads();
    }
    if (threadIdx.x == 0) atomicMax(amax, __float_as_uint(red[0]));
}

// ---------------------------------------------------------------------------
// 5) fake-quant + (optional RoPE) + repack to f16
//    mode 0: [S][E] -> [S][E]                (attention output)
//    mode 1: [S][E] -> [H][S][D] with RoPE   (q, k)
//    mode 2: [S][E] -> [H][D][S] transposed  (v; contiguous keys for PV B-op)
// ---------------------------------------------------------------------------
__global__ void quant_pack_kernel(const float* __restrict__ x, _Float16* __restrict__ dst,
                                  const unsigned* __restrict__ amax, int mode) {
    int idx = blockIdx.x * blockDim.x + threadIdx.x;  // S*H*32 threads
    int i = idx & 31;
    int h = (idx >> 5) & (N_HEAD - 1);
    int s = idx >> 10;
    if (s >= S_LEN) return;
    float scale = fmaxf(__uint_as_float(*amax) / 127.f, 1e-8f);
    size_t base = (size_t)s * E_DIM + (size_t)h * HEAD_D;
    float x1 = x[base + i];
    float x2 = x[base + i + 32];
    float q1 = fminf(fmaxf(rintf(x1 / scale), -128.f), 127.f) * scale;
    float q2 = fminf(fmaxf(rintf(x2 / scale), -128.f), 127.f) * scale;
    if (mode == 1) {
        float invf = __powf(10000.f, -(float)i / 32.f);
        float ang = (float)s * invf;
        float sn, cs;
        __sincosf(ang, &sn, &cs);
        float y1 = q1 * cs - q2 * sn;   // rotate_half: first half gets -x2
        float y2 = q2 * cs + q1 * sn;   // second half gets +x1
        q1 = y1; q2 = y2;
    }
    if (mode == 0) {
        dst[base + i]      = (_Float16)q1;
        dst[base + i + 32] = (_Float16)q2;
    } else if (mode == 2) {
        size_t ob = (size_t)h * HEAD_D * S_LEN + s;        // [h][d][s]
        dst[ob + (size_t)i * S_LEN]        = (_Float16)q1;
        dst[ob + (size_t)(i + 32) * S_LEN] = (_Float16)q2;
    } else {
        size_t ob = ((size_t)h * S_LEN + s) * HEAD_D;      // [h][s][d]
        dst[ob + i]      = (_Float16)q1;
        dst[ob + i + 32] = (_Float16)q2;
    }
}

// ---------------------------------------------------------------------------
// 6) causal flash attention, one wave (32 threads) per (head, 16-row block).
//    Q,K: [H][S][64] f16 ; V: [H][64][S] f16 ; O: [S][E] f32.
//    Pipelined: V-block loads issued at top of the iteration (hidden behind
//    score WMMAs + softmax VALU); next K-block fragments prefetched before
//    the softmax/LDS restage (clamped address, unconditional).
// ---------------------------------------------------------------------------
__global__ void __launch_bounds__(32) attn_kernel(const _Float16* __restrict__ Qh,
                                                  const _Float16* __restrict__ Kh,
                                                  const _Float16* __restrict__ Vt,
                                                  float* __restrict__ O) {
    __shared__ __align__(16) _Float16 pls[16 * 40];   // 16 rows x 32 cols, stride 40
    int wave = blockIdx.x;            // 32 heads * 128 row blocks
    int h  = wave >> 7;
    int ib = wave & 127;
    int lane = threadIdx.x;
    int half = lane >> 4;
    int ln   = lane & 15;
    int qbase = ib * 16;

    const _Float16* Q  = Qh + (size_t)h * S_LEN * HEAD_D;
    const _Float16* Kp = Kh + (size_t)h * S_LEN * HEAD_D;
    const _Float16* Vp = Vt + (size_t)h * HEAD_D * S_LEN;

    // Q tile in A-layout, split into two K=32 chunks (d 0..31 / 32..63)
    const _Float16* qrow = Q + (size_t)(qbase + ln) * HEAD_D;
    v16h aQ[2];
#pragma unroll
    for (int c = 0; c < 2; ++c) {
        v8h lo = *(const v8h*)(qrow + c * 32 + half * 16);
        v8h hi = *(const v8h*)(qrow + c * 32 + half * 16 + 8);
        aQ[c] = __builtin_shufflevector(lo, hi, 0,1,2,3,4,5,6,7,8,9,10,11,12,13,14,15);
    }

    float mrun[8], lrun[8];
#pragma unroll
    for (int r = 0; r < 8; ++r) { mrun[r] = -3.0e38f; lrun[r] = 0.f; }
    v8f accO[4] = {};

    int nkb = (qbase + 16 + 31) >> 5;   // key blocks of 32 (causal bound)

    // prologue: K fragments for block 0 ([t][d-chunk])
    v16h bK[2][2];
#pragma unroll
    for (int t = 0; t < 2; ++t) {
        const _Float16* krow = Kp + (size_t)(t * 16 + ln) * HEAD_D;
        bK[t][0] = *(const v16h*)(krow + half * 16);
        bK[t][1] = *(const v16h*)(krow + 32 + half * 16);
    }

    for (int kb = 0; kb < nkb; ++kb) {
        int kbase = kb * 32;
        // --- issue V loads early: consumed only at the end of the iteration ---
        v16h bV[4];
#pragma unroll
        for (int t = 0; t < 4; ++t)
            bV[t] = *(const v16h*)(Vp + (size_t)(t * 16 + ln) * S_LEN + kbase + half * 16);

        // --- scores: 16 rows x 32 keys, two 16-col subtiles ---
        v8f sc[2];
#pragma unroll
        for (int t = 0; t < 2; ++t) {
            v8f z = {};
            z = wmma_f16(aQ[0], bK[t][0], z);
            z = wmma_f16(aQ[1], bK[t][1], z);
            sc[t] = z;
        }

        // --- prefetch next K block (clamped; garbage unused on last iter) ---
        int kpre = kbase + 32 <= S_LEN - 32 ? kbase + 32 : S_LEN - 32;
        v16h nK[2][2];
#pragma unroll
        for (int t = 0; t < 2; ++t) {
            const _Float16* krow = Kp + (size_t)(kpre + t * 16 + ln) * HEAD_D;
            nK[t][0] = *(const v16h*)(krow + half * 16);
            nK[t][1] = *(const v16h*)(krow + 32 + half * 16);
        }

        // --- causal mask + online softmax (row r+8*half lives in element r) ---
        float pnew[2][8];
#pragma unroll
        for (int r = 0; r < 8; ++r) {
            int qrow_g = qbase + r + half * 8;
            float mloc = -3.0e38f;
#pragma unroll
            for (int t = 0; t < 2; ++t) {
                int key = kbase + t * 16 + ln;
                float v = (key <= qrow_g) ? sc[t][r] : -3.0e38f;
                sc[t][r] = v;
                mloc = fmaxf(mloc, v);
            }
            for (int msk = 1; msk < 16; msk <<= 1)
                mloc = fmaxf(mloc, __shfl_xor(mloc, msk, 32));
            float mnew = fmaxf(mrun[r], mloc);
            float corr = __expf(mrun[r] - mnew);
            float psum = 0.f;
#pragma unroll
            for (int t = 0; t < 2; ++t) {
                float p = __expf(sc[t][r] - mnew);   // masked -> exp(-huge) = 0
                pnew[t][r] = p;
                psum += p;
            }
            for (int msk = 1; msk < 16; msk <<= 1)
                psum += __shfl_xor(psum, msk, 32);
            lrun[r] = lrun[r] * corr + psum;
            mrun[r] = mnew;
#pragma unroll
            for (int t2 = 0; t2 < 4; ++t2) accO[t2][r] *= corr;
        }

        // --- restage P: D-layout -> LDS -> A-layout f16 ---
#pragma unroll
        for (int t = 0; t < 2; ++t)
#pragma unroll
            for (int r = 0; r < 8; ++r)
                pls[(r + half * 8) * 40 + t * 16 + ln] = (_Float16)pnew[t][r];
        asm volatile("s_wait_dscnt 0" ::: "memory");
        __builtin_amdgcn_wave_barrier();
        v8h plo = *(const v8h*)(&pls[ln * 40 + half * 16]);
        v8h phi = *(const v8h*)(&pls[ln * 40 + half * 16 + 8]);
        v16h aP = __builtin_shufflevector(plo, phi, 0,1,2,3,4,5,6,7,8,9,10,11,12,13,14,15);

        // --- PV: keys contiguous in Vt rows; bV loads issued at loop top ---
#pragma unroll
        for (int t = 0; t < 4; ++t)
            accO[t] = wmma_f16(aP, bV[t], accO[t]);

        // rotate K double buffer
#pragma unroll
        for (int t = 0; t < 2; ++t) { bK[t][0] = nK[t][0]; bK[t][1] = nK[t][1]; }
        __builtin_amdgcn_wave_barrier();  // keep LDS reuse ordered across iterations
    }

    // --- finalize: divide by row sums, write [S][E] ---
#pragma unroll
    for (int t = 0; t < 4; ++t)
#pragma unroll
        for (int r = 0; r < 8; ++r) {
            int mm = qbase + r + half * 8;
            int dd = t * 16 + ln;
            O[(size_t)mm * E_DIM + (size_t)h * HEAD_D + dd] = accO[t][r] / lrun[r];
        }
}

// ---------------------------------------------------------------------------
// launch
// ---------------------------------------------------------------------------
extern "C" void kernel_launch(void* const* d_in, const int* in_sizes, int n_in,
                              void* d_out, int out_size, void* d_ws, size_t ws_size,
                              hipStream_t stream) {
    (void)in_sizes; (void)n_in; (void)out_size; (void)ws_size;
    const float* hidden = (const float*)d_in[0];
    const float* w_q = (const float*)d_in[1];
    const float* w_k = (const float*)d_in[2];
    const float* w_v = (const float*)d_in[3];
    const float* w_o = (const float*)d_in[4];
    float* out = (float*)d_out;

    const size_t MB = 1ull << 20;
    char* w = (char*)d_ws;
    _Float16* wq_q = (_Float16*)(w + 0 * MB);
    _Float16* wq_k = (_Float16*)(w + 8 * MB);
    _Float16* wq_v = (_Float16*)(w + 16 * MB);
    _Float16* wq_o = (_Float16*)(w + 24 * MB);
    _Float16* h16  = (_Float16*)(w + 32 * MB);
    float* qf   = (float*)(w + 40 * MB);
    float* kf   = (float*)(w + 56 * MB);
    float* vf   = (float*)(w + 72 * MB);
    _Float16* Qh = (_Float16*)(w + 88 * MB);
    _Float16* Kh = (_Float16*)(w + 96 * MB);
    _Float16* Vt = (_Float16*)(w + 104 * MB);
    float* attn = (float*)(w + 112 * MB);
    _Float16* outq = (_Float16*)(w + 128 * MB);
    unsigned* amax = (unsigned*)(w + 136 * MB);  // [q, k, v, attn]

    const int NEL = S_LEN * E_DIM;               // 4M
    hipMemsetAsync(amax, 0, 4 * sizeof(unsigned), stream);

    cvt_f16_kernel<<<NEL / 256, 256, 0, stream>>>(hidden, h16, NEL);

    wquant_kernel<<<E_DIM, 256, 0, stream>>>(w_q, wq_q);
    wquant_kernel<<<E_DIM, 256, 0, stream>>>(w_k, wq_k);
    wquant_kernel<<<E_DIM, 256, 0, stream>>>(w_v, wq_v);
    wquant_kernel<<<E_DIM, 256, 0, stream>>>(w_o, wq_o);

    const float scaling = 0.125f;  // 64^-0.5
    gemm_wmma_kernel<<<512, 128, 0, stream>>>(h16, wq_q, qf, scaling);
    gemm_wmma_kernel<<<512, 128, 0, stream>>>(h16, wq_k, kf, 1.0f);
    gemm_wmma_kernel<<<512, 128, 0, stream>>>(h16, wq_v, vf, 1.0f);

    absmax_kernel<<<1024, 256, 0, stream>>>(qf, NEL, amax + 0);
    absmax_kernel<<<1024, 256, 0, stream>>>(kf, NEL, amax + 1);
    absmax_kernel<<<1024, 256, 0, stream>>>(vf, NEL, amax + 2);

    const int QP_THREADS = S_LEN * N_HEAD * 32;  // 2M (pair threads)
    quant_pack_kernel<<<QP_THREADS / 256, 256, 0, stream>>>(qf, Qh, amax + 0, 1);
    quant_pack_kernel<<<QP_THREADS / 256, 256, 0, stream>>>(kf, Kh, amax + 1, 1);
    quant_pack_kernel<<<QP_THREADS / 256, 256, 0, stream>>>(vf, Vt, amax + 2, 2);

    attn_kernel<<<N_HEAD * (S_LEN / 16), 32, 0, stream>>>(Qh, Kh, Vt, attn);

    absmax_kernel<<<1024, 256, 0, stream>>>(attn, NEL, amax + 3);
    quant_pack_kernel<<<QP_THREADS / 256, 256, 0, stream>>>(attn, outq, amax + 3, 0);

    gemm_wmma_kernel<<<512, 128, 0, stream>>>(outq, wq_o, out, 1.0f);
}
